// E71MatrixGatedCell_55456617726102
// MI455X (gfx1250) — matrix operand order repaired and checked by reading, compiled, not run on hardware
//
#include <hip/hip_runtime.h>
#include <hip/hip_bf16.h>

typedef __attribute__((ext_vector_type(16))) __bf16 v16bf;
typedef __attribute__((ext_vector_type(8)))  float  v8f;

#define T_STEPS 2048
#define BATCH   16
#define DIMK    1024
#define NDIM    64
#define ROWS    (T_STEPS * BATCH)   // 32768

// ---------------------------------------------------------------------------
// Convert 16 packed f32 (four float4) to a v16bf WMMA fragment register.
// ---------------------------------------------------------------------------
__device__ inline v16bf cvt16(float4 a, float4 b, float4 c, float4 d) {
  v16bf r;
  r[0]  = (__bf16)a.x; r[1]  = (__bf16)a.y; r[2]  = (__bf16)a.z; r[3]  = (__bf16)a.w;
  r[4]  = (__bf16)b.x; r[5]  = (__bf16)b.y; r[6]  = (__bf16)b.z; r[7]  = (__bf16)b.w;
  r[8]  = (__bf16)c.x; r[9]  = (__bf16)c.y; r[10] = (__bf16)c.z; r[11] = (__bf16)c.w;
  r[12] = (__bf16)d.x; r[13] = (__bf16)d.y; r[14] = (__bf16)d.z; r[15] = (__bf16)d.w;
  return r;
}

// ---------------------------------------------------------------------------
// Kernel A: fused 4-way projection GEMM  [32768 x 1024] x [1024 x 64] per W.
// Wave-per-tile: each wave computes a 64-row x 16-col tile of one projection.
// 512 row-blocks x 16 col-tiles (4 weights x 4 n-tiles) = 8192 waves.
// bf16 WMMA: D(16x16 f32) += A(16x32 bf16) x B(32x16 bf16), K-loop 32 iters.
// Inner body: load B + all 4 A fragments first, then 4 back-to-back WMMAs on
// disjoint accumulators -> no WMMA->VALU WAR hazard v_nops; hazard window is
// covered by the next iteration's global_load_b128s.
// ---------------------------------------------------------------------------
__global__ __launch_bounds__(256) void proj_gemm_kernel(
    const float* __restrict__ x,
    const float* __restrict__ Wk, const float* __restrict__ Wv,
    const float* __restrict__ Wq, const float* __restrict__ Wa,
    float* __restrict__ kp, float* __restrict__ vp,
    float* __restrict__ qp, float* __restrict__ ap)
{
  const int tid  = threadIdx.x;
  const int wid  = blockIdx.x * 8 + (tid >> 5);   // 0..8191
  const int rb   = wid & 511;                     // row block of 64 rows
  const int ct   = wid >> 9;                      // 0..15
  const int wsel = ct >> 2;                       // which weight matrix
  const int n0   = (ct & 3) * 16;                 // output column base

  const float* W = (wsel == 0) ? Wk : (wsel == 1) ? Wv : (wsel == 2) ? Wq : Wa;
  float*       O = (wsel == 0) ? kp : (wsel == 1) ? vp : (wsel == 2) ? qp : ap;

  const int lane = tid & 31;
  const int p    = lane & 15;      // row-in-tile (A) / col (B,D)
  const int h    = lane >> 4;      // half-wave selector

  v8f acc[4];
  acc[0] = (v8f){}; acc[1] = (v8f){}; acc[2] = (v8f){}; acc[3] = (v8f){};

  const int    row_base = rb * 64;
  const float* wrow     = W + (size_t)(n0 + p) * DIMK + h * 8;
  const float* xbase    = x + (size_t)(row_base + p) * DIMK + h * 8;

  for (int k0 = 0; k0 < DIMK; k0 += 32) {
    // B fragment: lane holds K values of W row (n0+p), half-wave h covers
    // K offsets [h*8, h*8+8) and [16+h*8, 16+h*8+8) within the 32-wide block.
    const float4* bp = (const float4*)(wrow + k0);
    v16bf bfrag = cvt16(bp[0], bp[1], bp[4], bp[5]);

    // All four A fragments up front (rows row_base + s*16 + p).
    v16bf afrag[4];
#pragma unroll
    for (int s = 0; s < 4; ++s) {
      const float* xr = xbase + (size_t)s * 16 * DIMK + k0;
      float4 a0 = *(const float4*)(xr);
      float4 a1 = *(const float4*)(xr + 4);
      float4 a2 = *(const float4*)(xr + 16);
      float4 a3 = *(const float4*)(xr + 20);
      afrag[s] = cvt16(a0, a1, a2, a3);
    }

    // Back-to-back WMMAs: disjoint accumulators, shared B fragment.
#pragma unroll
    for (int s = 0; s < 4; ++s) {
      acc[s] = __builtin_amdgcn_wmma_f32_16x16x32_bf16(
          /*neg_a=*/false, afrag[s], /*neg_b=*/false, bfrag,
          /*c_mod=*/(short)0, acc[s], /*reuse_a=*/false, /*reuse_b=*/false);
    }
  }

  // D layout: element r of lane l -> row (r + 8*h), col p   (16x16 f32 C/D map)
#pragma unroll
  for (int s = 0; s < 4; ++s) {
#pragma unroll
    for (int r = 0; r < 8; ++r) {
      int row = row_base + s * 16 + r + 8 * h;
      O[(size_t)row * NDIM + n0 + p] = acc[s][r];
    }
  }
}

// ---------------------------------------------------------------------------
// Kernel B: sequential gated-state scan. One workgroup per batch (16 blocks).
// 256 threads: 4 lanes per state row, 16 columns per lane, S fully in VGPRs.
// Row-dot reductions via 2x shfl_xor (wave32). Projections are L2-resident;
// prefetch next step while computing the current one.
// ---------------------------------------------------------------------------
__global__ __launch_bounds__(256) void scan_kernel(
    const float* __restrict__ Sin,
    const float* __restrict__ d_alpha, const float* __restrict__ b_alpha,
    const float* __restrict__ kp, const float* __restrict__ vp,
    const float* __restrict__ qp, const float* __restrict__ ap,
    float* __restrict__ out, float* __restrict__ Sfin)
{
  const int b   = blockIdx.x;
  const int tid = threadIdx.x;
  const int i   = tid >> 2;        // state row 0..63
  const int c   = tid & 3;         // column group
  const int c0  = c * 16;

  float S[16];
  {
    const float* sp = Sin + ((size_t)b * NDIM + i) * NDIM + c0;
#pragma unroll
    for (int j = 0; j < 16; j += 4) {
      float4 t = *(const float4*)(sp + j);
      S[j] = t.x; S[j + 1] = t.y; S[j + 2] = t.z; S[j + 3] = t.w;
    }
  }

  const float da = d_alpha[i];
  const float ba = b_alpha[i];

  for (int t = 0; t < T_STEPS; ++t) {
    const int idx = (t * BATCH + b) * NDIM;

    if (t + 1 < T_STEPS) {                       // hide next step's L2 latency
      const int nidx = idx + BATCH * NDIM;
      __builtin_prefetch(kp + nidx + c0, 0, 0);
      __builtin_prefetch(qp + nidx + c0, 0, 0);
      __builtin_prefetch(vp + nidx + i, 0, 0);
      __builtin_prefetch(ap + nidx + i, 0, 0);
    }

    float kk[16], qq[16];
#pragma unroll
    for (int j = 0; j < 16; j += 4) {
      float4 tk = *(const float4*)(kp + idx + c0 + j);
      kk[j] = tk.x; kk[j + 1] = tk.y; kk[j + 2] = tk.z; kk[j + 3] = tk.w;
      float4 tq = *(const float4*)(qp + idx + c0 + j);
      qq[j] = tq.x; qq[j + 1] = tq.y; qq[j + 2] = tq.z; qq[j + 3] = tq.w;
    }

    // retrieved_i = sum_j S[i,j] * k[j]
    float r = 0.0f;
#pragma unroll
    for (int j = 0; j < 16; ++j) r = fmaf(S[j], kk[j], r);
    r += __shfl_xor(r, 1, 32);
    r += __shfl_xor(r, 2, 32);

    const float ai = ap[idx + i];
    const float vi = vp[idx + i];
    const float z  = ai + da * r + ba;
    const float alpha = 1.0f / (1.0f + __expf(-z));
    const float om    = (1.0f - alpha) * vi;

    // S = alpha*S + (1-alpha)*v k^T ;  o = S . q
    float o = 0.0f;
#pragma unroll
    for (int j = 0; j < 16; ++j) {
      S[j] = fmaf(alpha, S[j], om * kk[j]);
      o    = fmaf(S[j], qq[j], o);
    }
    o += __shfl_xor(o, 1, 32);
    o += __shfl_xor(o, 2, 32);

    if (c == 0) {
      const float sig = 1.0f / (1.0f + __expf(-o));
      out[idx + i] = o * o * sig;              // o * silu(o)
    }
  }

  float* fp = Sfin + ((size_t)b * NDIM + i) * NDIM + c0;
#pragma unroll
  for (int j = 0; j < 16; ++j) fp[j] = S[j];
}

// ---------------------------------------------------------------------------
extern "C" void kernel_launch(void* const* d_in, const int* in_sizes, int n_in,
                              void* d_out, int out_size, void* d_ws, size_t ws_size,
                              hipStream_t stream) {
  const float* x  = (const float*)d_in[0];
  const float* S0 = (const float*)d_in[1];
  const float* Wk = (const float*)d_in[2];
  const float* Wv = (const float*)d_in[3];
  const float* Wq = (const float*)d_in[4];
  const float* Wa = (const float*)d_in[5];
  const float* da = (const float*)d_in[6];
  const float* ba = (const float*)d_in[7];

  float* ws = (float*)d_ws;                    // needs 4*32768*64*4 = 32 MB
  float* kp = ws;
  float* vp = ws + (size_t)ROWS * NDIM;
  float* qp = ws + (size_t)2 * ROWS * NDIM;
  float* ap = ws + (size_t)3 * ROWS * NDIM;

  float* out  = (float*)d_out;                        // [T,B,N]
  float* Sfin = out + (size_t)T_STEPS * BATCH * NDIM; // [B,N,N]

  proj_gemm_kernel<<<1024, 256, 0, stream>>>(x, Wk, Wv, Wq, Wa, kp, vp, qp, ap);
  scan_kernel<<<BATCH, 256, 0, stream>>>(S0, da, ba, kp, vp, qp, ap, out, Sfin);
}
